// Head_4286377361590
// MI455X (gfx1250) — matrix operand order repaired and checked by reading, not compiled, not run on hardware
//
#include <hip/hip_runtime.h>
#include <hip/hip_bf16.h>

#define BB 512
#define TT 256
#define CC 384
#define HH 64
#define WN (HH * CC)   // 24576 elements per weight matrix

#if __has_builtin(__builtin_amdgcn_global_load_async_to_lds_b128)
#define HAS_ASYNC_LDS 1
#else
#define HAS_ASYNC_LDS 0
#endif

#if __has_builtin(__builtin_amdgcn_s_wait_asynccnt)
#define WAIT_ASYNC(n) __builtin_amdgcn_s_wait_asynccnt(n)
#else
#define WAIT_ASYNC(n) asm volatile("s_wait_asynccnt %0" ::"i"(n) : "memory")
#endif

typedef __attribute__((ext_vector_type(16))) __bf16 bf16x16;
typedef __attribute__((ext_vector_type(8)))  float  f32x8;

union BFV {
    bf16x16 v;
    uint4   q[2];
};

__device__ __forceinline__ f32x8 wmma_bf16(bf16x16 a, bf16x16 b, f32x8 c) {
    // (neg_a, A, neg_b, B, c_mod, C, reuse_a, reuse_b)
    return __builtin_amdgcn_wmma_f32_16x16x32_bf16(false, a, false, b, (short)0, c, false, false);
}

__device__ __forceinline__ void put4(float* f, float4 v) {
    f[0] = v.x; f[1] = v.y; f[2] = v.z; f[3] = v.w;
}

__device__ __forceinline__ bf16x16 cvt16(const float* f) {
    BFV r;
#pragma unroll
    for (int i = 0; i < 16; ++i) r.v[i] = (__bf16)f[i];
    return r.v;
}

// 16 contiguous f32 -> bf16x16 (inline-convert fallback path)
__device__ __forceinline__ bf16x16 load_w16(const float* p) {
    float f[16];
    put4(f + 0,  *(const float4*)(p + 0));
    put4(f + 4,  *(const float4*)(p + 4));
    put4(f + 8,  *(const float4*)(p + 8));
    put4(f + 12, *(const float4*)(p + 12));
    return cvt16(f);
}

// 16 contiguous bf16 (pre-converted weights)
__device__ __forceinline__ bf16x16 load_wb16(const __bf16* p) {
    BFV w;
    w.q[0] = *(const uint4*)(p);
    w.q[1] = *(const uint4*)(p + 8);
    return w.v;
}

#if HAS_ASYNC_LDS
typedef int v4i __attribute__((vector_size(16)));
typedef __attribute__((address_space(1))) v4i v4i_g;   // global (AS1)
typedef __attribute__((address_space(3))) v4i v4i_l;   // LDS (AS3)

// Async-copy one 16-row x 32-col f32 x-chunk into a per-wave LDS buffer.
// 4 instructions x 32 lanes x 16B = 2 KB.
__device__ __forceinline__ void issue_x_chunk(const float* xt, int c0, float* buf, int lane) {
#pragma unroll
    for (int i = 0; i < 4; ++i) {
        const int row = i * 4 + (lane >> 3);
        const int seg = lane & 7;
        const float* g = xt + (size_t)row * CC + c0 + seg * 4;
        float*       d = buf + row * 32 + seg * 4;
        __builtin_amdgcn_global_load_async_to_lds_b128(
            (v4i_g*)(unsigned long long)g,   // AS1: integer reinterpret (value-identical)
            (v4i_l*)(v4i*)d,                 // AS3: explicit address-space conversion
            0, 0);
    }
}
#endif

// One-shot f32 -> bf16 weight conversion into workspace: layout [Wk | Wq | Wv].
__global__ __launch_bounds__(256)
void prep_weights(const float* __restrict__ Wk, const float* __restrict__ Wq,
                  const float* __restrict__ Wv, __bf16* __restrict__ wbf) {
    const int idx = (blockIdx.x * 256 + threadIdx.x) * 4;   // 72 blocks cover 3*WN
    const int m   = idx / WN;
    const int off = idx % WN;
    const float* src = (m == 0) ? Wk : (m == 1) ? Wq : Wv;
    const float4 v = *(const float4*)(src + off);
    __bf16* d = wbf + idx;
    d[0] = (__bf16)v.x; d[1] = (__bf16)v.y; d[2] = (__bf16)v.z; d[3] = (__bf16)v.w;
}

template <bool PREPPED>
__global__ __launch_bounds__(256)
void attn_head_kernel(const float* __restrict__ x,
                      const float* __restrict__ Wk_f,
                      const float* __restrict__ Wq_f,
                      const float* __restrict__ Wv_f,
                      const __bf16* __restrict__ wkb,
                      const __bf16* __restrict__ wqb,
                      const __bf16* __restrict__ wvb,
                      float* __restrict__ out) {
    __shared__ __bf16 Ks[TT * HH];        // K  as [key][h]   (32 KB)
    __shared__ __bf16 Vt[HH * TT];        // V^T as [h][key]  (32 KB)
    __shared__ __bf16 Scr[8][16 * HH];    // per-wave staging (16 KB)
#if HAS_ASYNC_LDS
    __shared__ float  Xb[8][2][16 * 32];  // per-wave x double buffer (32 KB)
#endif

    const int b    = blockIdx.x;
    const int tid  = threadIdx.x;
    const int wave = tid >> 5;
    const int lane = tid & 31;
    const int hi   = lane >> 4;    // which 16-lane half
    const int l16  = lane & 15;

    __bf16* scr = &Scr[wave][0];

    bf16x16 qa[2][2];   // Q fragments (A-layout), 2 tiles x 2 h-chunks

    // ---------------- Phase 1: fused QKV projection ----------------
#pragma unroll
    for (int ti = 0; ti < 2; ++ti) {
        const int rt   = (ti == 0) ? wave : (15 - wave);
        const int row0 = rt * 16;

        f32x8 aq[4], ak[4], av[4];
#pragma unroll
        for (int n0 = 0; n0 < 4; ++n0) {
            aq[n0] = (f32x8){0,0,0,0,0,0,0,0};
            ak[n0] = (f32x8){0,0,0,0,0,0,0,0};
            av[n0] = (f32x8){0,0,0,0,0,0,0,0};
        }

#if HAS_ASYNC_LDS
        float* const xb0 = &Xb[wave][0][0];
        float* const xb1 = &Xb[wave][1][0];
        const float* xt = x + ((size_t)b * TT + row0) * CC;
        issue_x_chunk(xt, 0, xb0, lane);
#else
        const float* xrow = x + ((size_t)b * TT + row0 + l16) * CC;
#endif

        for (int c0 = 0; c0 < CC; c0 += 32) {
            float af[16];
#if HAS_ASYNC_LDS
            const int cb   = (c0 >> 5) & 1;
            float* cur = cb ? xb1 : xb0;
            float* nxt = cb ? xb0 : xb1;
            if (c0 + 32 < CC) {
                issue_x_chunk(xt, c0 + 32, nxt, lane);
                WAIT_ASYNC(4);     // in-order completion: previous chunk landed
            } else {
                WAIT_ASYNC(0);
            }
            const float* xr = cur + l16 * 32;
            put4(af + 0,  *(const float4*)(xr + hi * 16));
            put4(af + 4,  *(const float4*)(xr + hi * 16 + 4));
            put4(af + 8,  *(const float4*)(xr + hi * 16 + 8));
            put4(af + 12, *(const float4*)(xr + hi * 16 + 12));
#else
            __builtin_prefetch(xrow + c0 + 64, 0, 0);
            put4(af + 0,  *(const float4*)(xrow + c0 + hi * 16));
            put4(af + 4,  *(const float4*)(xrow + c0 + hi * 16 + 4));
            put4(af + 8,  *(const float4*)(xrow + c0 + hi * 16 + 8));
            put4(af + 12, *(const float4*)(xrow + c0 + hi * 16 + 12));
#endif
            const bf16x16 a = cvt16(af);

#pragma unroll
            for (int n0 = 0; n0 < 4; ++n0) {
                const int h = n0 * 16 + l16;
                const size_t wo = (size_t)h * CC + c0 + hi * 16;
                bf16x16 bq, bk, bv;
                if constexpr (PREPPED) {
                    bq = load_wb16(wqb + wo);
                    bk = load_wb16(wkb + wo);
                    bv = load_wb16(wvb + wo);
                } else {
                    bq = load_w16(Wq_f + wo);
                    bk = load_w16(Wk_f + wo);
                    bv = load_w16(Wv_f + wo);
                }
                aq[n0] = wmma_bf16(a, bq, aq[n0]);
                ak[n0] = wmma_bf16(a, bk, ak[n0]);
                av[n0] = wmma_bf16(a, bv, av[n0]);
            }
        }

        // scatter results: K -> Ks[key][h], V -> Vt[h][key], Q -> wave scratch
#pragma unroll
        for (int n0 = 0; n0 < 4; ++n0) {
#pragma unroll
            for (int r = 0; r < 8; ++r) {
                const int key = row0 + r + 8 * hi;   // C-layout row
                const int h   = n0 * 16 + l16;       // C-layout col
                Ks[key * HH + h]           = (__bf16)ak[n0][r];
                Vt[h * TT + key]           = (__bf16)av[n0][r];
                scr[(r + 8 * hi) * HH + h] = (__bf16)aq[n0][r];
            }
        }
        // re-load Q from scratch in A-layout (per-lane row = l16)
#pragma unroll
        for (int ch = 0; ch < 2; ++ch) {
            BFV qv;
            qv.q[0] = *(const uint4*)(scr + l16 * HH + ch * 32 + hi * 16);
            qv.q[1] = *(const uint4*)(scr + l16 * HH + ch * 32 + hi * 16 + 8);
            qa[ti][ch] = qv.v;
        }
    }

    __syncthreads();

    // ---------------- Phase 2: causal flash attention ----------------
#pragma unroll
    for (int ti = 0; ti < 2; ++ti) {
        const int qt   = (ti == 0) ? wave : (15 - wave);
        const int row0 = qt * 16;

        float mi[8], li[8];
#pragma unroll
        for (int r = 0; r < 8; ++r) { mi[r] = -1e30f; li[r] = 0.0f; }
        f32x8 o[4];
#pragma unroll
        for (int n0 = 0; n0 < 4; ++n0) o[n0] = (f32x8){0,0,0,0,0,0,0,0};

        const int kend = (qt + 1) * 16;
        for (int kb = 0; kb < kend; kb += 32) {
            // scores S = Q K^T for 32 keys (two 16x16 tiles)
            f32x8 s[2];
#pragma unroll
            for (int t = 0; t < 2; ++t) {
                const int key = kb + t * 16 + l16;   // B-layout col = key
                f32x8 acc = (f32x8){0,0,0,0,0,0,0,0};
#pragma unroll
                for (int ch = 0; ch < 2; ++ch) {
                    BFV kv;
                    const __bf16* kp = &Ks[key * HH + ch * 32 + hi * 16];
                    kv.q[0] = *(const uint4*)(kp);
                    kv.q[1] = *(const uint4*)(kp + 8);
                    acc = wmma_bf16(qa[ti][ch], kv.v, acc);
                }
                s[t] = acc;
            }

            // scale + causal mask + online softmax (row = r + 8*hi of tile)
            float pv[2][8], alpha[8];
#pragma unroll
            for (int r = 0; r < 8; ++r) {
                const int row = row0 + r + 8 * hi;
                float v0 = s[0][r] * 0.125f;
                float v1 = s[1][r] * 0.125f;
                if (kb + l16      > row) v0 = -1e30f;
                if (kb + 16 + l16 > row) v1 = -1e30f;
                float m = fmaxf(v0, v1);
#pragma unroll
                for (int d = 1; d < 16; d <<= 1) m = fmaxf(m, __shfl_xor(m, d, 32));
                const float mn = fmaxf(mi[r], m);
                alpha[r] = __expf(mi[r] - mn);
                mi[r] = mn;
                const float e0 = __expf(v0 - mn);
                const float e1 = __expf(v1 - mn);
                pv[0][r] = e0; pv[1][r] = e1;
                float rs = e0 + e1;
#pragma unroll
                for (int d = 1; d < 16; d <<= 1) rs += __shfl_xor(rs, d, 32);
                li[r] = li[r] * alpha[r] + rs;
            }
#pragma unroll
            for (int n0 = 0; n0 < 4; ++n0)
#pragma unroll
                for (int r = 0; r < 8; ++r) o[n0][r] *= alpha[r];

            // P (16x32) C-layout -> bf16 A-layout via wave scratch
#pragma unroll
            for (int t = 0; t < 2; ++t)
#pragma unroll
                for (int r = 0; r < 8; ++r)
                    scr[(r + 8 * hi) * 32 + t * 16 + l16] = (__bf16)pv[t][r];

            BFV pa;
            pa.q[0] = *(const uint4*)(scr + l16 * 32 + hi * 16);
            pa.q[1] = *(const uint4*)(scr + l16 * 32 + hi * 16 + 8);

            // O += P V  (B-tiles from Vt: per-lane 16 consecutive keys)
#pragma unroll
            for (int n0 = 0; n0 < 4; ++n0) {
                BFV vb;
                const __bf16* vp = &Vt[(n0 * 16 + l16) * TT + kb + hi * 16];
                vb.q[0] = *(const uint4*)(vp);
                vb.q[1] = *(const uint4*)(vp + 8);
                o[n0] = wmma_bf16(pa.v, vb.v, o[n0]);
            }
        }

        // normalize + store
        float* ob = out + ((size_t)b * TT + row0) * HH;
#pragma unroll
        for (int n0 = 0; n0 < 4; ++n0)
#pragma unroll
            for (int r = 0; r < 8; ++r)
                ob[(size_t)(r + 8 * hi) * HH + n0 * 16 + l16] = o[n0][r] / li[r];
    }
}

extern "C" void kernel_launch(void* const* d_in, const int* in_sizes, int n_in,
                              void* d_out, int out_size, void* d_ws, size_t ws_size,
                              hipStream_t stream) {
    const float* x  = (const float*)d_in[0];
    const float* Wk = (const float*)d_in[1];
    const float* Wq = (const float*)d_in[2];
    const float* Wv = (const float*)d_in[3];
    float* out = (float*)d_out;

    dim3 grid(BB);
    dim3 block(256);

    const size_t need = (size_t)3 * WN * sizeof(unsigned short);
    if (d_ws != nullptr && ws_size >= need) {
        __bf16* wbf = (__bf16*)d_ws;
        prep_weights<<<dim3(3 * WN / (256 * 4)), block, 0, stream>>>(Wk, Wq, Wv, wbf);
        attn_head_kernel<true><<<grid, block, 0, stream>>>(
            x, Wk, Wq, Wv, wbf, wbf + WN, wbf + 2 * WN, out);
    } else {
        attn_head_kernel<false><<<grid, block, 0, stream>>>(
            x, Wk, Wq, Wv, nullptr, nullptr, nullptr, out);
    }
}
